// FuzzySelfAttention_31542239822476
// MI455X (gfx1250) — matrix operand order repaired and checked by reading, not compiled, not run on hardware
//
#include <hip/hip_runtime.h>

// MI455X / gfx1250, wave32. All GEMMs via v_wmma_f32_16x16x32_f16.
// B=8, S=2048, D=512 (fixed by the reference).

typedef __attribute__((ext_vector_type(16))) _Float16 v16h;
typedef __attribute__((ext_vector_type(8)))  _Float16 v8h;
typedef __attribute__((ext_vector_type(4)))  _Float16 v4h;
typedef __attribute__((ext_vector_type(8)))  float    v8f;

constexpr int Bn = 8;
constexpr int Sn = 2048;
constexpr int Dn = 512;

union V16U { v16h v; v8h h[2]; };

// A-fragment (M=16 x K=32, f16), row-major source with `stride` halves per row.
// ISA layout: lanes 0-15: M=lane, halves = K{0..7,16..23}; lanes 16-31: M=lane-16,
// halves = K{8..15,24..31}.
__device__ __forceinline__ v16h load_a_frag(const _Float16* base, int stride, int lane) {
  const int ln = lane & 15;
  const int lg = (lane >> 4) & 1;
  const _Float16* p = base + ln * stride + lg * 8;
  V16U u;
  u.h[0] = *(const v8h*)(p);
  u.h[1] = *(const v8h*)(p + 16);
  return u.v;
}

// B-fragment (K=32 x N=16, f16): N=lane%16, halves = K{0..7,16..23} (lanes 0-15) / K{8..15,24..31} (lanes 16-31).
__device__ __forceinline__ v16h load_b_frag(const _Float16* base, int stride, int lane) {
  const int n = lane & 15;
  const int h = (lane >> 4) & 1;
  v16h r;
#pragma unroll
  for (int s = 0; s < 16; ++s) {
    const int k = (s < 8) ? (8 * h + s) : (16 + 8 * h + (s - 8));
    r[s] = base[k * stride + n];
  }
  return r;
}

__device__ __forceinline__ v8f wmma_f16(v16h a, v16h b, v8f c) {
  return __builtin_amdgcn_wmma_f32_16x16x32_f16(false, a, false, b, (short)0, c,
                                                false, false);
}

// Score tile: 16 queries x 16 keys, K-dim = 512 (16 WMMA steps), depth-2
// software-pipelined B-fragment loads so WMMA latency covers the L2 loads.
__device__ __forceinline__ v8f score_tile(const v16h qf[16], const _Float16* Kcol,
                                          int lane) {
  v8f s = {};
  v16h b0 = load_b_frag(Kcol + 0 * 32 * Sn, Sn, lane);
  v16h b1 = load_b_frag(Kcol + 1 * 32 * Sn, Sn, lane);
#pragma unroll
  for (int dc = 0; dc < 16; ++dc) {
    const int dn = (dc + 2 < 16) ? dc + 2 : 15;          // constant-folded clamp
    const v16h bn = load_b_frag(Kcol + dn * 32 * Sn, Sn, lane);
    s = wmma_f16(qf[dc], b0, s);
    b0 = b1;
    b1 = bn;
  }
  return s;
}

// ---------------------------------------------------------------- convert x -> f16
__global__ void cvt_x_kernel(const float* __restrict__ x, _Float16* __restrict__ xh) {
  const int i = blockIdx.x * blockDim.x + threadIdx.x;  // exactly B*S*D/4 threads
  const float4 v = ((const float4*)x)[i];
  v4h o;
  o[0] = (_Float16)v.x; o[1] = (_Float16)v.y;
  o[2] = (_Float16)v.z; o[3] = (_Float16)v.w;
  ((v4h*)xh)[i] = o;
}

// --------------------------------------------- transpose+convert W[e,d] -> Wt[d,e] f16
__global__ void cvt_w_kernel(const float* __restrict__ Wq, const float* __restrict__ Wk,
                             _Float16* __restrict__ WqT, _Float16* __restrict__ WkT) {
  const int t = blockIdx.x * blockDim.x + threadIdx.x;  // exactly D*D threads
  const int e = t >> 9;
  const int d = t & (Dn - 1);
  WqT[d * Dn + e] = (_Float16)Wq[t];
  WkT[d * Dn + e] = (_Float16)Wk[t];
}

// ------------------------------------------------- projection: Q = xWq^T+bq, K = xWk^T+bk
// Q stored [b,s,e] f16; K stored TRANSPOSED [b,e,s] f16 so that the score-GEMM
// B-fragment (lane = key) reads consecutive keys across lanes.
// Fully staged pipeline: A + 8 B fragments for iteration dc+1 are loaded before
// the 8 WMMAs of iteration dc issue.
__global__ __launch_bounds__(256) void proj_kernel(
    const _Float16* __restrict__ Xh, const _Float16* __restrict__ WqT,
    const _Float16* __restrict__ WkT, const float* __restrict__ bq,
    const float* __restrict__ bk, _Float16* __restrict__ Qh,
    _Float16* __restrict__ KhT) {
  const int lane = threadIdx.x & 31;
  const int wave = threadIdx.x >> 5;
  const int gw = blockIdx.x * 8 + wave;   // 8192 waves
  const int mt = gw >> 3;                  // row tile 0..1023
  const int es = gw & 7;                   // e strip 0..7 (64 cols each)
  const int row0 = mt * 16;
  const int e0 = es * 64;

  v8f zero = {};
  v8f accq[4], acck[4];
#pragma unroll
  for (int j = 0; j < 4; ++j) { accq[j] = zero; acck[j] = zero; }

  v16h a_cur = load_a_frag(Xh + row0 * Dn, Dn, lane);
  v16h bq_cur[4], bk_cur[4];
#pragma unroll
  for (int j = 0; j < 4; ++j) {
    bq_cur[j] = load_b_frag(WqT + e0 + j * 16, Dn, lane);
    bk_cur[j] = load_b_frag(WkT + e0 + j * 16, Dn, lane);
  }

#pragma unroll 1
  for (int dc = 0; dc < 16; ++dc) {
    const int dn = (dc + 1 < 16) ? dc + 1 : 15;
    const int d1 = dn * 32;
    const v16h a_nxt = load_a_frag(Xh + row0 * Dn + d1, Dn, lane);
    v16h bq_nxt[4], bk_nxt[4];
#pragma unroll
    for (int j = 0; j < 4; ++j) {
      bq_nxt[j] = load_b_frag(WqT + d1 * Dn + e0 + j * 16, Dn, lane);
      bk_nxt[j] = load_b_frag(WkT + d1 * Dn + e0 + j * 16, Dn, lane);
    }
#pragma unroll
    for (int j = 0; j < 4; ++j) {
      accq[j] = wmma_f16(a_cur, bq_cur[j], accq[j]);
      acck[j] = wmma_f16(a_cur, bk_cur[j], acck[j]);
    }
    a_cur = a_nxt;
#pragma unroll
    for (int j = 0; j < 4; ++j) { bq_cur[j] = bq_nxt[j]; bk_cur[j] = bk_nxt[j]; }
  }

  const int ln = lane & 15, lg = lane >> 4;
#pragma unroll
  for (int j = 0; j < 4; ++j) {
    const int eg = e0 + j * 16 + ln;
    const float bqv = bq[eg];
    const float bkv = bk[eg];
#pragma unroll
    for (int r = 0; r < 8; ++r) {
      const int grow = row0 + r + 8 * lg;     // global row in [0, B*S)
      Qh[grow * Dn + eg] = (_Float16)(accq[j][r] + bqv);
      const int bb = grow >> 11;              // / S
      const int ss = grow & (Sn - 1);
      KhT[(bb * Dn + eg) * Sn + ss] = (_Float16)(acck[j][r] + bkv);
    }
  }
}

// ----------------------------------------------------------- fused fuzzy attention
// One wave owns a 16-query tile. Pass 1: online softmax stats (m, l) over all keys.
// Pass 2: recompute scores, p = fuzzy(exp(s-m)/l), LDS re-layout C/D->A fragment,
// accumulate P @ X with WMMA into 16x512 f32 registers.
__global__ __launch_bounds__(256) void attn_kernel(
    const _Float16* __restrict__ Xh, const _Float16* __restrict__ Qh,
    const _Float16* __restrict__ KhT, float* __restrict__ out) {
  __shared__ __attribute__((aligned(32))) _Float16 pbuf[8][16][32];

  const int lane = threadIdx.x & 31;
  const int wave = threadIdx.x >> 5;
  const int tile = blockIdx.x * 8 + wave;   // 1024 tiles
  const int b = tile >> 7;
  const int qrow0 = (tile & 127) * 16;

  const _Float16* Qbase = Qh + (b * Sn + qrow0) * Dn;
  const _Float16* Kbase = KhT + b * Dn * Sn;
  const _Float16* Xbase = Xh + b * Sn * Dn;
  float* Obase = out + (b * Sn + qrow0) * Dn;

  const int lg = lane >> 4;
  const int ln = lane & 15;

  // Cache the 16 Q A-fragments (whole 16x512 Q tile) in VGPRs.
  v16h qf[16];
#pragma unroll
  for (int dc = 0; dc < 16; ++dc) qf[dc] = load_a_frag(Qbase + dc * 32, Dn, lane);

  float mrun[8], lrun[8];
#pragma unroll
  for (int r = 0; r < 8; ++r) { mrun[r] = -3.0e38f; lrun[r] = 0.0f; }

  // -------- Pass 1: row max + sum of exp (online), scores via WMMA --------
#pragma unroll 1
  for (int kt = 0; kt < 128; ++kt) {
    const v8f s = score_tile(qf, Kbase + kt * 16, lane);
#pragma unroll
    for (int r = 0; r < 8; ++r) {
      const float v = s[r];
      float tmax = v;
#pragma unroll
      for (int msk = 1; msk < 16; msk <<= 1)
        tmax = fmaxf(tmax, __shfl_xor(tmax, msk, 32));
      const float mo = mrun[r];
      const float mn = fmaxf(mo, tmax);
      float e = __expf(v - mn);
#pragma unroll
      for (int msk = 1; msk < 16; msk <<= 1) e += __shfl_xor(e, msk, 32);
      lrun[r] = lrun[r] * __expf(mo - mn) + e;
      mrun[r] = mn;
    }
  }
  float linv[8];
#pragma unroll
  for (int r = 0; r < 8; ++r) linv[r] = 1.0f / lrun[r];

  v8f zero = {};
  v8f acc[32];
#pragma unroll
  for (int nt = 0; nt < 32; ++nt) acc[nt] = zero;

  const float inv03 = 1.0f / 0.3f;

  // -------- Pass 2: recompute scores, fuzzy(p), P @ V via WMMA --------
#pragma unroll 1
  for (int kc = 0; kc < 64; ++kc) {
#pragma unroll
    for (int half = 0; half < 2; ++half) {
      const int kcol = kc * 32 + half * 16;
      const v8f s = score_tile(qf, Kbase + kcol, lane);
#pragma unroll
      for (int r = 0; r < 8; ++r) {
        const float p = __expf(s[r] - mrun[r]) * linv[r];
        const float fm = fmaxf(fminf(p * inv03, (1.0f - p) * inv03), 0.0f);
        pbuf[wave][r + 8 * lg][half * 16 + ln] = (_Float16)fm;
      }
    }
    __syncthreads();  // uniform; orders LDS writes vs A-fragment reads

    const v16h ap = load_a_frag(&pbuf[wave][0][0], 32, lane);
    const int krow0 = kc * 32;
    if (kc < 63)  // prefetch next chunk's V rows (global_prefetch_b8)
      __builtin_prefetch(Xbase + (krow0 + 32 + lane) * Dn, 0, 1);

    const _Float16* Vrow = Xbase + krow0 * Dn;
    v16h b0 = load_b_frag(Vrow + 0 * 16, Dn, lane);
    v16h b1 = load_b_frag(Vrow + 1 * 16, Dn, lane);
#pragma unroll
    for (int nt = 0; nt < 32; ++nt) {
      const int nn = (nt + 2 < 32) ? nt + 2 : 31;        // constant-folded clamp
      const v16h bnv = load_b_frag(Vrow + nn * 16, Dn, lane);
      acc[nt] = wmma_f16(ap, b0, acc[nt]);
      b0 = b1;
      b1 = bnv;
    }
    __syncthreads();
  }

  // -------- store 16x512 f32 output tile --------
#pragma unroll
  for (int nt = 0; nt < 32; ++nt) {
#pragma unroll
    for (int r = 0; r < 8; ++r)
      Obase[(r + 8 * lg) * Dn + nt * 16 + ln] = acc[nt][r];
  }
}

extern "C" void kernel_launch(void* const* d_in, const int* in_sizes, int n_in,
                              void* d_out, int out_size, void* d_ws, size_t ws_size,
                              hipStream_t stream) {
  const float* x  = (const float*)d_in[0];
  const float* Wq = (const float*)d_in[1];
  const float* bq = (const float*)d_in[2];
  const float* Wk = (const float*)d_in[3];
  const float* bk = (const float*)d_in[4];
  float* out = (float*)d_out;

  // Workspace layout (f16): Xh | Qh | KhT | WqT | WkT  (~51 MB total)
  _Float16* Xh  = (_Float16*)d_ws;
  _Float16* Qh  = Xh + (size_t)Bn * Sn * Dn;
  _Float16* KhT = Qh + (size_t)Bn * Sn * Dn;
  _Float16* WqT = KhT + (size_t)Bn * Sn * Dn;
  _Float16* WkT = WqT + (size_t)Dn * Dn;

  cvt_x_kernel<<<(Bn * Sn * Dn / 4) / 256, 256, 0, stream>>>(x, Xh);
  cvt_w_kernel<<<(Dn * Dn) / 256, 256, 0, stream>>>(Wq, Wk, WqT, WkT);
  proj_kernel<<<1024, 256, 0, stream>>>(Xh, WqT, WkT, bq, bk, Qh, KhT);
  attn_kernel<<<128, 256, 0, stream>>>(Xh, Qh, KhT, out);
}
